// FastReadLayer_51256139710507
// MI455X (gfx1250) — matrix operand order repaired and checked by reading, not compiled, not run on hardware
//
#include <hip/hip_runtime.h>
#include <math.h>

typedef __attribute__((ext_vector_type(8)))  float  v8f;
typedef __attribute__((ext_vector_type(8)))  __bf16 v8bf;
typedef __attribute__((ext_vector_type(16))) __bf16 v16bf;

// ---------------------------------------------------------------------------
// bf16x3 helpers: x = hi + lo (each RN bf16) -> ~16 mantissa bits.
// ---------------------------------------------------------------------------
__device__ __forceinline__ unsigned short f2bf_rn(float x) {
    unsigned u = __float_as_uint(x);
    unsigned r = u + 0x7FFFu + ((u >> 16) & 1u);
    return (unsigned short)(r >> 16);
}
__device__ __forceinline__ float bf2f(unsigned short h) {
    return __uint_as_float(((unsigned)h) << 16);
}
__device__ __forceinline__ void splitbf(float x, unsigned short& h, unsigned short& l) {
    h = f2bf_rn(x);
    l = f2bf_rn(x - bf2f(h));
}

// D = A(16x32 bf16) * B(32x16 bf16) + C(16x16 f32), wave32.
__device__ __forceinline__ v8f wmma_bf(v16bf a, v16bf b, v8f c) {
    return __builtin_amdgcn_wmma_f32_16x16x32_bf16(false, a, false, b,
                                                   (short)0, c, false, false);
}
__device__ __forceinline__ v16bf cat8(v8bf a, v8bf b) {
    return __builtin_shufflevector(a, b, 0,1,2,3,4,5,6,7,8,9,10,11,12,13,14,15);
}
// A frag (16x32, MxK): lane l, row m=l&15, h=l>>4: K-runs [8h,8h+8) and [16+8h,..+8)
__device__ __forceinline__ v16bf fragA(const unsigned short* rowp, int hh) {
    return cat8(*(const v8bf*)(rowp + 8 * hh), *(const v8bf*)(rowp + 16 + 8 * hh));
}
// B frag (32x16, KxN): lane l, col n=l&15, h=l>>4: K-runs [8h,8h+8) and [16+8h,..+8)
__device__ __forceinline__ v16bf fragB(const unsigned short* rowp, int hh) {
    return cat8(*(const v8bf*)(rowp + 8 * hh), *(const v8bf*)(rowp + 16 + 8 * hh));
}

// ===========================================================================
// Generic split kernel: fp32 -> (hi, lo) bf16 planes
// ===========================================================================
__global__ __launch_bounds__(256)
void split_convert(const float* __restrict__ in, unsigned short* __restrict__ hi,
                   unsigned short* __restrict__ lo, int n) {
    int i = blockIdx.x * 256 + threadIdx.x;
    if (i < n) { unsigned short h, l; splitbf(in[i], h, l); hi[i] = h; lo[i] = l; }
}

// ===========================================================================
// Transpose + split: xhr (B,256,4096) fp32 -> T (B,4096,256) bf16 hi/lo
// ===========================================================================
__global__ __launch_bounds__(256)
void transpose_split_xhr(const float* __restrict__ xhr,
                         unsigned short* __restrict__ Thi,
                         unsigned short* __restrict__ Tlo) {
    __shared__ float tile[32][33];
    const int b = blockIdx.z, n0 = blockIdx.x * 32, k0 = blockIdx.y * 32;
    const int tx = threadIdx.x & 31, ty = threadIdx.x >> 5;   // 32 x 8
    const float* src = xhr + ((long long)b * 256 + k0) * 4096 + n0;
#pragma unroll
    for (int i = 0; i < 32; i += 8) tile[ty + i][tx] = src[(long long)(ty + i) * 4096 + tx];
    __syncthreads();
    unsigned short* dh = Thi + ((long long)b * 4096 + n0) * 256 + k0;
    unsigned short* dl = Tlo + ((long long)b * 4096 + n0) * 256 + k0;
#pragma unroll
    for (int i = 0; i < 32; i += 8) {
        unsigned short h, l;
        splitbf(tile[tx][ty + i], h, l);        // T[n0+ty+i][k0+tx] = xhr[k0+tx][n0+ty+i]
        dh[(ty + i) * 256 + tx] = h;
        dl[(ty + i) * 256 + tx] = l;
    }
}

// ===========================================================================
// Unified bf16x3 NT GEMM: C[b][m][n] = sum_k A[b][m][k] * Bt[b][n][k] (+ epi)
// A, Bt pre-split into bf16 hi/lo planes, row-major along k.
// Block 256 thr (8 waves), tile 64(M) x 128(N) x 32(K); wave 32x32 (2x2 WMMA),
// 12 x v_wmma_f32_16x16x32_bf16 per chunk (hh + hl + lh terms).
// Double-buffered LDS: stage chunk i+1 while computing chunk i (1 barrier/chunk)
// + global_prefetch of chunk i+2.
// EPI: 0 = /rowsum -> fp32   (attention read)
//      1 = +bias[n]+add -> fp32 (residual)
//      2 = gelu(acc+bias[n]) -> bf16 hi/lo planes
//      3 = logits/values: m<512 -> fp32 +bias[m]; m>=512 -> split(+bias2) planes
// ===========================================================================
template <int EPI>
__global__ __launch_bounds__(256)
void gemm_bf3(const unsigned short* __restrict__ Ahi, const unsigned short* __restrict__ Alo,
              long long strideA, int lda,
              const unsigned short* __restrict__ Bhi, const unsigned short* __restrict__ Blo,
              long long strideB, int ldb,
              float* __restrict__ C, int ldc,
              unsigned short* __restrict__ OutHi, unsigned short* __restrict__ OutLo,
              const float* __restrict__ bias, const float* __restrict__ bias2,
              const float* __restrict__ add, const float* __restrict__ rowsum,
              int M, int N, int K) {
    constexpr int LD = 40;  // 32 k + pad: 80B rows, 16B aligned, conflict-free b128
    __shared__ __align__(16) unsigned short AsH[2][64 * LD],  AsL[2][64 * LD];
    __shared__ __align__(16) unsigned short BsH[2][128 * LD], BsL[2][128 * LD];

    const int b  = blockIdx.z;
    const int m0 = blockIdx.y * 64;
    const int n0 = blockIdx.x * 128;
    const unsigned short* AbH = Ahi + (long long)b * strideA;
    const unsigned short* AbL = Alo + (long long)b * strideA;
    const unsigned short* BbH = Bhi + (long long)b * strideB;
    const unsigned short* BbL = Blo + (long long)b * strideB;

    const int tid  = threadIdx.x;
    const int lane = tid & 31;
    const int wave = tid >> 5;
    const int wm   = (wave >> 2) * 32;
    const int wn   = (wave & 3) * 32;
    const int hh   = lane >> 4;
    const int ml   = lane & 15;

    const int ra = tid >> 2, qa = (tid & 3) * 8;    // A stage: 8 ushorts / thread / plane
    const int rb = tid >> 1, qb = (tid & 1) * 16;   // B stage: 16 ushorts / thread / plane
    const long long offA0 = (long long)(m0 + ra) * lda + qa;
    const long long offB0 = (long long)(n0 + rb) * ldb + qb;

    auto stage = [&](int k0, int buf) {
        const long long oa = offA0 + k0;
        *(uint4*)&AsH[buf][ra * LD + qa] = *(const uint4*)(AbH + oa);
        *(uint4*)&AsL[buf][ra * LD + qa] = *(const uint4*)(AbL + oa);
        const long long ob = offB0 + k0;
        *(uint4*)&BsH[buf][rb * LD + qb]     = *(const uint4*)(BbH + ob);
        *(uint4*)&BsH[buf][rb * LD + qb + 8] = *(const uint4*)(BbH + ob + 8);
        *(uint4*)&BsL[buf][rb * LD + qb]     = *(const uint4*)(BbL + ob);
        *(uint4*)&BsL[buf][rb * LD + qb + 8] = *(const uint4*)(BbL + ob + 8);
    };

    v8f c00 = {}, c01 = {}, c10 = {}, c11 = {};
    const int nchunks = K >> 5;

    stage(0, 0);
    __syncthreads();

    for (int chunk = 0; chunk < nchunks; ++chunk) {
        const int cur = chunk & 1;
        if (chunk + 1 < nchunks) stage((chunk + 1) << 5, cur ^ 1);
        if (chunk + 2 < nchunks) {   // prefetch chunk+2 into cache hierarchy
            const int kp = (chunk + 2) << 5;
            __builtin_prefetch((const void*)(AbH + offA0 + kp), 0, 3);
            __builtin_prefetch((const void*)(BbH + offB0 + kp), 0, 3);
            __builtin_prefetch((const void*)(AbL + offA0 + kp), 0, 3);
            __builtin_prefetch((const void*)(BbL + offB0 + kp), 0, 3);
        }

        v16bf ah0 = fragA(&AsH[cur][(wm + ml) * LD], hh);
        v16bf ah1 = fragA(&AsH[cur][(wm + 16 + ml) * LD], hh);
        v16bf al0 = fragA(&AsL[cur][(wm + ml) * LD], hh);
        v16bf al1 = fragA(&AsL[cur][(wm + 16 + ml) * LD], hh);
        v16bf bh0 = fragB(&BsH[cur][(wn + ml) * LD], hh);
        v16bf bh1 = fragB(&BsH[cur][(wn + 16 + ml) * LD], hh);
        v16bf bl0 = fragB(&BsL[cur][(wn + ml) * LD], hh);
        v16bf bl1 = fragB(&BsL[cur][(wn + 16 + ml) * LD], hh);

        c00 = wmma_bf(ah0, bh0, c00);
        c01 = wmma_bf(ah0, bh1, c01);
        c10 = wmma_bf(ah1, bh0, c10);
        c11 = wmma_bf(ah1, bh1, c11);
        c00 = wmma_bf(ah0, bl0, c00);
        c01 = wmma_bf(ah0, bl1, c01);
        c10 = wmma_bf(ah1, bl0, c10);
        c11 = wmma_bf(ah1, bl1, c11);
        c00 = wmma_bf(al0, bh0, c00);
        c01 = wmma_bf(al0, bh1, c01);
        c10 = wmma_bf(al1, bh0, c10);
        c11 = wmma_bf(al1, bh1, c11);

        __syncthreads();   // nxt fully staged; cur fully consumed
    }

    const int nlo  = lane & 15;
    const int mhi8 = (lane >> 4) * 8;
    v8f accs[4] = {c00, c01, c10, c11};
#pragma unroll
    for (int t = 0; t < 4; ++t) {
        const int mb = m0 + wm + (t >> 1) * 16 + mhi8;
        const int n  = n0 + wn + (t & 1) * 16 + nlo;
#pragma unroll
        for (int p = 0; p < 8; ++p) {
            const int m = mb + p;
            const float v = accs[t][p];
            if (EPI == 0) {
                C[((long long)b * M + m) * (long long)ldc + n] = v / rowsum[b * M + m];
            } else if (EPI == 1) {
                const long long idx = ((long long)b * M + m) * (long long)ldc + n;
                C[idx] = v + bias[n] + add[idx];
            } else if (EPI == 2) {
                const float x = v + bias[n];
                const float g = 0.5f * x * (1.0f + erff(x * 0.70710678118654752f));
                unsigned short h, l;
                splitbf(g, h, l);
                const long long idx = ((long long)b * M + m) * (long long)ldc + n;
                OutHi[idx] = h; OutLo[idx] = l;
            } else {  // EPI == 3: logits (fp32) / values (bf16 planes)
                if (m < 512) {
                    C[((long long)b * 512 + m) * 4096 + n] = v + bias[m];
                } else {
                    unsigned short h, l;
                    splitbf(v + bias2[m - 512], h, l);
                    const long long idx = ((long long)b * 512 + (m - 512)) * 4096 + n;
                    OutHi[idx] = h; OutLo[idx] = l;
                }
            }
        }
    }
}

// ===========================================================================
// Top-k threshold + masked unnormalized softmax.
// Reads fp32 logits row; writes exp values as bf16 hi/lo planes EMBEDDED in
// the (now dead) fp32 row: hi -> first 8KB, lo -> second 8KB.
// ===========================================================================
__device__ __forceinline__ unsigned f2key(float x) {
    unsigned u = __float_as_uint(x);
    return (u & 0x80000000u) ? ~u : (u | 0x80000000u);
}
__device__ __forceinline__ float key2f(unsigned key) {
    unsigned u = (key & 0x80000000u) ? (key & 0x7FFFFFFFu) : ~key;
    return __uint_as_float(u);
}

__global__ __launch_bounds__(256)
void topk_softmax_kernel(float* __restrict__ PVlog, float* __restrict__ rowsum) {
    __shared__ unsigned keys[4096];
    __shared__ unsigned hist[256];
    __shared__ unsigned sscan[256];
    __shared__ unsigned sel_s, selk_s;
    __shared__ unsigned ured[256];
    __shared__ float    fred[256];

    const int row = blockIdx.x;                          // b*512 + l
    float* p = PVlog + (long long)row * 4096;
    const int tid = threadIdx.x;

    unsigned mymax = 0u;
    for (int i = tid; i < 4096; i += 256) {
        const unsigned key = f2key(p[i]);
        keys[i] = key;
        mymax = (key > mymax) ? key : mymax;
    }
    ured[tid] = mymax;
    __syncthreads();
    for (int off = 128; off > 0; off >>= 1) {
        if (tid < off) ured[tid] = (ured[tid + off] > ured[tid]) ? ured[tid + off] : ured[tid];
        __syncthreads();
    }
    const float rowmax = key2f(ured[0]);

    unsigned prefix = 0u, kth = 256u;
    for (int shift = 24; shift >= 0; shift -= 8) {
        hist[tid] = 0u;
        __syncthreads();
        const unsigned hmask = (shift == 24) ? 0u : (0xFFFFFFFFu << (shift + 8));
        for (int i = tid; i < 4096; i += 256) {
            const unsigned key = keys[i];
            if ((key & hmask) == prefix)
                atomicAdd(&hist[(key >> shift) & 255u], 1u);
        }
        __syncthreads();
        sscan[tid] = hist[tid];
        __syncthreads();
        for (int off = 1; off < 256; off <<= 1) {       // suffix sums cnt_ge[t]
            const unsigned v = (tid + off < 256) ? sscan[tid + off] : 0u;
            __syncthreads();
            sscan[tid] += v;
            __syncthreads();
        }
        const unsigned ge  = sscan[tid];
        const unsigned gen = (tid < 255) ? sscan[tid + 1] : 0u;
        if (ge >= kth && gen < kth) { sel_s = (unsigned)tid; selk_s = kth - gen; }
        __syncthreads();
        prefix |= (sel_s << shift);
        kth = selk_s;
        __syncthreads();
    }
    const unsigned threshKey = prefix;

    unsigned short* eh = (unsigned short*)p;   // 4096 hi in first 2048 floats
    unsigned short* el = eh + 4096;            // 4096 lo in second 2048 floats
    float lsum = 0.f;
    for (int i = tid; i < 4096; i += 256) {
        const unsigned key = keys[i];
        float e = 0.f;
        if (key >= threshKey) e = expf(key2f(key) - rowmax);
        unsigned short h, l;
        splitbf(e, h, l);
        eh[i] = h; el[i] = l;
        lsum += e;
    }
    fred[tid] = lsum;
    __syncthreads();
    for (int off = 128; off > 0; off >>= 1) {
        if (tid < off) fred[tid] += fred[tid + off];
        __syncthreads();
    }
    if (tid == 0) rowsum[row] = fred[0];
}

// ===========================================================================
// LN over concat(xlr_row[512], read_row[512]) -> hn bf16 hi/lo (1024 wide)
// ===========================================================================
__global__ __launch_bounds__(256)
void ln_concat_kernel(const float* __restrict__ xlr, const float* __restrict__ rd,
                      const float* __restrict__ g, const float* __restrict__ be,
                      unsigned short* __restrict__ oh, unsigned short* __restrict__ ol) {
    const int row = blockIdx.x;
    const int tid = threadIdx.x;
    const float* a = xlr + (long long)row * 512;
    const float* r = rd  + (long long)row * 512;
    __shared__ float red[256];

    const float v0 = a[tid], v1 = a[tid + 256], v2 = r[tid], v3 = r[tid + 256];
    red[tid] = v0 + v1 + v2 + v3;
    __syncthreads();
    for (int off = 128; off > 0; off >>= 1) {
        if (tid < off) red[tid] += red[tid + off];
        __syncthreads();
    }
    const float mu = red[0] * (1.0f / 1024.0f);
    __syncthreads();
    const float d0 = v0 - mu, d1 = v1 - mu, d2 = v2 - mu, d3 = v3 - mu;
    red[tid] = d0 * d0 + d1 * d1 + d2 * d2 + d3 * d3;
    __syncthreads();
    for (int off = 128; off > 0; off >>= 1) {
        if (tid < off) red[tid] += red[tid + off];
        __syncthreads();
    }
    const float rs = rsqrtf(red[0] * (1.0f / 1024.0f) + 1e-5f);
    unsigned short* ph = oh + (long long)row * 1024;
    unsigned short* pl = ol + (long long)row * 1024;
    const float dd[4] = {d0, d1, d2, d3};
#pragma unroll
    for (int j = 0; j < 4; ++j) {
        const int i = tid + j * 256;
        unsigned short h, l;
        splitbf(dd[j] * rs * g[i] + be[i], h, l);
        ph[i] = h; pl[i] = l;
    }
}

// LN over 512-wide rows -> zn bf16 hi/lo
__global__ __launch_bounds__(256)
void ln_kernel(const float* __restrict__ x, const float* __restrict__ g,
               const float* __restrict__ be,
               unsigned short* __restrict__ oh, unsigned short* __restrict__ ol) {
    const int row = blockIdx.x;
    const int tid = threadIdx.x;
    const float* p = x + (long long)row * 512;
    __shared__ float red[256];

    const float v0 = p[tid], v1 = p[tid + 256];
    red[tid] = v0 + v1;
    __syncthreads();
    for (int off = 128; off > 0; off >>= 1) {
        if (tid < off) red[tid] += red[tid + off];
        __syncthreads();
    }
    const float mu = red[0] * (1.0f / 512.0f);
    __syncthreads();
    const float d0 = v0 - mu, d1 = v1 - mu;
    red[tid] = d0 * d0 + d1 * d1;
    __syncthreads();
    for (int off = 128; off > 0; off >>= 1) {
        if (tid < off) red[tid] += red[tid + off];
        __syncthreads();
    }
    const float rs = rsqrtf(red[0] * (1.0f / 512.0f) + 1e-5f);
    unsigned short* ph = oh + (long long)row * 512;
    unsigned short* pl = ol + (long long)row * 512;
    unsigned short h, l;
    splitbf(d0 * rs * g[tid] + be[tid], h, l);
    ph[tid] = h; pl[tid] = l;
    splitbf(d1 * rs * g[tid + 256] + be[tid + 256], h, l);
    ph[tid + 256] = h; pl[tid + 256] = l;
}

// ===========================================================================
extern "C" void kernel_launch(void* const* d_in, const int* in_sizes, int n_in,
                              void* d_out, int out_size, void* d_ws, size_t ws_size,
                              hipStream_t stream) {
    const float* xlr   = (const float*)d_in[0];
    const float* xhr   = (const float*)d_in[1];
    const float* Wlog  = (const float*)d_in[2];
    const float* blog  = (const float*)d_in[3];
    const float* Wval  = (const float*)d_in[4];
    const float* bval  = (const float*)d_in[5];
    const float* g1    = (const float*)d_in[6];
    const float* be1   = (const float*)d_in[7];
    const float* Wproj = (const float*)d_in[8];
    const float* bproj = (const float*)d_in[9];
    const float* g2    = (const float*)d_in[10];
    const float* be2   = (const float*)d_in[11];
    const float* W1    = (const float*)d_in[12];
    const float* b1    = (const float*)d_in[13];
    const float* W2    = (const float*)d_in[14];
    const float* b2    = (const float*)d_in[15];
    float* out = (float*)d_out;
    char* base = (char*)d_ws;

    // ---- workspace layout (byte offsets) ----
    // [0, 134.2MB): PVlog fp32 (16x512x4096) w/ embedded E hi/lo after top-k;
    //               later reused for hn/zn/a1 bf16 planes (all dead by then).
    float*          PVlog = (float*)base;
    unsigned short* hnHi  = (unsigned short*)(base);
    unsigned short* hnLo  = (unsigned short*)(base + 16777216);
    unsigned short* znHi  = (unsigned short*)(base + 33554432);
    unsigned short* znLo  = (unsigned short*)(base + 41943040);
    unsigned short* a1Hi  = (unsigned short*)(base + 50331648);
    unsigned short* a1Lo  = (unsigned short*)(base + 83886080);
    unsigned short* Vhi   = (unsigned short*)(base + 134217728);   // 67.1MB
    unsigned short* Vlo   = (unsigned short*)(base + 201326592);   // 67.1MB
    unsigned short* Thi   = (unsigned short*)(base + 268435456);   // 33.5MB (xhrT)
    unsigned short* Tlo   = (unsigned short*)(base + 301989888);   // 33.5MB
    float*          readb = (float*)(base + 268435456);            // alias Thi (dead)
    float*          xlr2  = (float*)(base + 285212672);            // alias Thi (dead)
    unsigned short* WcH   = (unsigned short*)(base + 335544320);   // Wcat 1024x256
    unsigned short* WcL   = (unsigned short*)(base + 336068608);
    unsigned short* WpH   = (unsigned short*)(base + 336592896);   // Wproj 512x1024
    unsigned short* WpL   = (unsigned short*)(base + 337641472);
    unsigned short* W1H   = (unsigned short*)(base + 338690048);   // W1 2048x512
    unsigned short* W1L   = (unsigned short*)(base + 340787200);
    unsigned short* W2H   = (unsigned short*)(base + 342884352);   // W2 512x2048
    unsigned short* W2L   = (unsigned short*)(base + 344981504);
    float*          rowsum= (float*)(base + 347078656);

    const dim3 blk(256);

    // 0) split all weights once (bf16x3 planes)
    split_convert<<<dim3(512),  blk, 0, stream>>>(Wlog,  WcH,           WcL,           131072);
    split_convert<<<dim3(512),  blk, 0, stream>>>(Wval,  WcH + 131072,  WcL + 131072,  131072);
    split_convert<<<dim3(2048), blk, 0, stream>>>(Wproj, WpH,           WpL,           524288);
    split_convert<<<dim3(4096), blk, 0, stream>>>(W1,    W1H,           W1L,           1048576);
    split_convert<<<dim3(4096), blk, 0, stream>>>(W2,    W2H,           W2L,           1048576);

    // 0b) transpose + split xhr -> T (B,4096,256)
    transpose_split_xhr<<<dim3(128, 8, 16), blk, 0, stream>>>(xhr, Thi, Tlo);

    // 1) logits (fp32) + values (bf16 planes):  Wcat @ xhr[b]
    gemm_bf3<3><<<dim3(32, 16, 16), blk, 0, stream>>>(
        WcH, WcL, 0, 256, Thi, Tlo, 4096LL * 256, 256,
        PVlog, 4096, Vhi, Vlo, blog, bval, nullptr, nullptr, 1024, 4096, 256);

    // 2) exact top-256 threshold + masked exp (writes E hi/lo in place) + rowsum
    topk_softmax_kernel<<<dim3(16 * 512), blk, 0, stream>>>(PVlog, rowsum);

    // 3) read = (E @ V^T) / rowsum
    gemm_bf3<0><<<dim3(4, 8, 16), blk, 0, stream>>>(
        (unsigned short*)PVlog, (unsigned short*)PVlog + 4096, 512LL * 8192, 8192,
        Vhi, Vlo, 512LL * 4096, 4096,
        readb, 512, nullptr, nullptr, nullptr, nullptr, nullptr, rowsum, 512, 512, 4096);

    // 4) hn = LN(concat(xlr, read)) * g1 + be1   (bf16 planes)
    ln_concat_kernel<<<dim3(16 * 512), blk, 0, stream>>>(xlr, readb, g1, be1, hnHi, hnLo);

    // 5) xlr2 = xlr + hn @ Wproj^T + bproj
    gemm_bf3<1><<<dim3(4, 8, 16), blk, 0, stream>>>(
        hnHi, hnLo, 512LL * 1024, 1024, WpH, WpL, 0, 1024,
        xlr2, 512, nullptr, nullptr, bproj, nullptr, xlr, nullptr, 512, 512, 1024);

    // 6) zn = LN(xlr2) * g2 + be2   (bf16 planes)
    ln_kernel<<<dim3(16 * 512), blk, 0, stream>>>(xlr2, g2, be2, znHi, znLo);

    // 7) a1 = gelu(zn @ W1^T + b1)  (bf16 planes)
    gemm_bf3<2><<<dim3(16, 8, 16), blk, 0, stream>>>(
        znHi, znLo, 512LL * 512, 512, W1H, W1L, 0, 512,
        nullptr, 2048, a1Hi, a1Lo, b1, nullptr, nullptr, nullptr, 512, 2048, 512);

    // 8) out = xlr2 + a1 @ W2^T + b2
    gemm_bf3<1><<<dim3(4, 8, 16), blk, 0, stream>>>(
        a1Hi, a1Lo, 512LL * 2048, 2048, W2H, W2L, 0, 2048,
        out, 512, nullptr, nullptr, b2, nullptr, xlr2, nullptr, 512, 512, 2048);
}
